// GCNForecast_37426345017425
// MI455X (gfx1250) — matrix operand order repaired and checked by reading, not compiled, not run on hardware
//
#include <hip/hip_runtime.h>
#include <hip/hip_bf16.h>

typedef __attribute__((ext_vector_type(16))) __bf16 v16bf;
typedef __attribute__((ext_vector_type(8)))  float  v8f;

#define HIDDEN 128

// ---------------- degree / normalization ----------------
__global__ void k_deg_init(float* deg, int N) {
  int i = blockIdx.x * blockDim.x + threadIdx.x;
  if (i < N) deg[i] = 1.0f;                       // self-loop contributes 1
}

__global__ void k_deg_accum(const int* __restrict__ col, float* deg, int E) {
  int e = blockIdx.x * blockDim.x + threadIdx.x;
  if (e < E) unsafeAtomicAdd(&deg[col[e]], 1.0f); // global_atomic_add_f32
}

__global__ void k_rsqrt_inplace(float* deg, int N) {
  int i = blockIdx.x * blockDim.x + threadIdx.x;
  if (i < N) deg[i] = rsqrtf(deg[i]);             // deg >= 1 always
}

// ---------------- WMMA GEMM:  C[nrows x 128] = A[nrows x 128] @ W[128 x 128] ----------------
// 256 threads = 8 waves; each wave produces one 16-row tile of C.
// W is packed once per block into LDS in the exact v_wmma B-operand lane layout.
__global__ __launch_bounds__(256) void k_gemm_bf16(
    const float* __restrict__ A, const float* __restrict__ W,
    float* __restrict__ C, int nrows, int ntiles)
{
  __shared__ alignas(32) __bf16 wlds[4 * 8 * 32 * 16];   // [kt][nt][lane][j]  = 32 KB
  const int tid = threadIdx.x;

  // Pack: wlds[((kt*8+nt)*32 + lane)*16 + j] = bf16( W[(kt*32 + (lane>>4)*8 + (j&7) + (j>>3)*16)*128 + nt*16 + (lane&15)] )
  #pragma unroll 4
  for (int t = 0; t < 64; ++t) {
    int idx  = tid * 64 + t;
    int j    =  idx        & 15;
    int lane = (idx >> 4)  & 31;
    int nt   = (idx >> 9)  & 7;
    int kt   =  idx >> 12;
    int k    = kt * 32 + (lane >> 4) * 8 + (j & 7) + (j >> 3) * 16;
    wlds[idx] = (__bf16)W[k * HIDDEN + nt * 16 + (lane & 15)];
  }
  __syncthreads();

  const int wave = tid >> 5;
  const int lane = tid & 31;
  const int tile = blockIdx.x * 8 + wave;
  if (tile >= ntiles) return;                      // wave-uniform: EXEC stays all-ones

  const int half = lane >> 4;
  const int m    = lane & 15;
  int row = tile * 16 + m;
  if (row >= nrows) row = nrows - 1;               // load clamp only (N%16==0 in practice)

  v8f acc[8] = {};                                 // 8 N-tiles of 16 cols

  #pragma unroll
  for (int kt = 0; kt < 4; ++kt) {
    // A operand: lane holds K in [8*half, 8*half+8) and [16+8*half, 16+8*half+8)
    const float* ap = A + (size_t)row * HIDDEN + kt * 32 + half * 8;
    float4 c0 = *(const float4*)(ap +  0);
    float4 c1 = *(const float4*)(ap +  4);
    float4 c2 = *(const float4*)(ap + 16);
    float4 c3 = *(const float4*)(ap + 20);
    v16bf a;
    a[0]  = (__bf16)c0.x; a[1]  = (__bf16)c0.y; a[2]  = (__bf16)c0.z; a[3]  = (__bf16)c0.w;
    a[4]  = (__bf16)c1.x; a[5]  = (__bf16)c1.y; a[6]  = (__bf16)c1.z; a[7]  = (__bf16)c1.w;
    a[8]  = (__bf16)c2.x; a[9]  = (__bf16)c2.y; a[10] = (__bf16)c2.z; a[11] = (__bf16)c2.w;
    a[12] = (__bf16)c3.x; a[13] = (__bf16)c3.y; a[14] = (__bf16)c3.z; a[15] = (__bf16)c3.w;

    #pragma unroll
    for (int nt = 0; nt < 8; ++nt) {
      v16bf b = *(const v16bf*)&wlds[(((kt * 8) + nt) * 32 + lane) * 16];
      acc[nt] = __builtin_amdgcn_wmma_f32_16x16x32_bf16(
          false, a, false, b, (short)0, acc[nt], false, false);
    }
  }

  // D layout: VGPR r -> row (r + 8*half), col m
  #pragma unroll
  for (int nt = 0; nt < 8; ++nt) {
    #pragma unroll
    for (int r = 0; r < 8; ++r) {
      int orow = tile * 16 + half * 8 + r;
      if (orow < nrows) C[(size_t)orow * HIDDEN + nt * 16 + m] = acc[nt][r];
    }
  }
}

// ---------------- self-loop init: AGG[i] = dinv[i]^2 * H[i]  (full overwrite) ----------------
__global__ void k_selfloop_init(const float* __restrict__ dinv, const float* __restrict__ H,
                                float* __restrict__ AGG, int N) {
  int gid = blockIdx.x * blockDim.x + threadIdx.x;
  int i = gid >> 5;
  if (i >= N) return;
  int c = (gid & 31) * 4;
  float d = dinv[i];
  float s = d * d;
  const float4 h = *(const float4*)(H + (size_t)i * HIDDEN + c);
  float4 o; o.x = s * h.x; o.y = s * h.y; o.z = s * h.z; o.w = s * h.w;
  *(float4*)(AGG + (size_t)i * HIDDEN + c) = o;
}

// ---------------- edge scatter: AGG[col] += dinv[row]*dinv[col] * H[row] ----------------
// one wave per edge, one float4 column chunk per lane
__global__ void k_edge_scatter(const int* __restrict__ row, const int* __restrict__ col,
                               const float* __restrict__ dinv, const float* __restrict__ H,
                               float* __restrict__ AGG, int E) {
  int gid = blockIdx.x * blockDim.x + threadIdx.x;
  int e = gid >> 5;
  if (e >= E) return;
  int c  = (gid & 31) * 4;
  int r  = row[e];
  int co = col[e];
  float nrm = dinv[r] * dinv[co];
  const float4 h = *(const float4*)(H + (size_t)r * HIDDEN + c);
  float* dst = AGG + (size_t)co * HIDDEN + c;
  unsafeAtomicAdd(dst + 0, nrm * h.x);
  unsafeAtomicAdd(dst + 1, nrm * h.y);
  unsafeAtomicAdd(dst + 2, nrm * h.z);
  unsafeAtomicAdd(dst + 3, nrm * h.w);
}

// ---------------- bias + relu, in place ----------------
__global__ void k_bias_relu(float* __restrict__ H, const float* __restrict__ b, int N) {
  int gid = blockIdx.x * blockDim.x + threadIdx.x;
  int i = gid >> 5;
  if (i >= N) return;
  int c = (gid & 31) * 4;
  float4 v = *(const float4*)(H + (size_t)i * HIDDEN + c);
  v.x = fmaxf(v.x + b[c + 0], 0.0f);
  v.y = fmaxf(v.y + b[c + 1], 0.0f);
  v.z = fmaxf(v.z + b[c + 2], 0.0f);
  v.w = fmaxf(v.w + b[c + 3], 0.0f);
  *(float4*)(H + (size_t)i * HIDDEN + c) = v;
}

// ---------------- pooling ----------------
__global__ void k_zero(float* p, int n) {
  int i = blockIdx.x * blockDim.x + threadIdx.x;
  if (i < n) p[i] = 0.0f;
}

__global__ void k_pool_accum(const int* __restrict__ batch, const float* __restrict__ H,
                             float* __restrict__ pooled, float* __restrict__ cnt, int N) {
  int gid = blockIdx.x * blockDim.x + threadIdx.x;
  int i = gid >> 5;
  if (i >= N) return;
  int c = (gid & 31) * 4;
  int g = batch[i];
  const float4 h = *(const float4*)(H + (size_t)i * HIDDEN + c);
  float* dst = pooled + (size_t)g * HIDDEN + c;
  unsafeAtomicAdd(dst + 0, h.x);
  unsafeAtomicAdd(dst + 1, h.y);
  unsafeAtomicAdd(dst + 2, h.z);
  unsafeAtomicAdd(dst + 3, h.w);
  if ((gid & 31) == 0) unsafeAtomicAdd(&cnt[g], 1.0f);
}

// out[g] = dot(pooled_sum[g], lin_w) / max(cnt[g],1) + lin_b
__global__ __launch_bounds__(128) void k_head(const float* __restrict__ pooled,
                                              const float* __restrict__ cnt,
                                              const float* __restrict__ lin_w,
                                              const float* __restrict__ lin_b,
                                              float* __restrict__ out) {
  __shared__ float red[128];
  int g = blockIdx.x;
  int t = threadIdx.x;
  red[t] = pooled[(size_t)g * HIDDEN + t] * lin_w[t];
  __syncthreads();
  #pragma unroll
  for (int s = 64; s > 0; s >>= 1) {
    if (t < s) red[t] += red[t + s];
    __syncthreads();
  }
  if (t == 0) {
    float c = cnt[g];
    c = c > 1.0f ? c : 1.0f;
    out[g] = red[0] / c + lin_b[0];
  }
}

static inline size_t align256(size_t x) { return (x + 255) & ~(size_t)255; }

extern "C" void kernel_launch(void* const* d_in, const int* in_sizes, int n_in,
                              void* d_out, int out_size, void* d_ws, size_t ws_size,
                              hipStream_t stream) {
  const float* x     = (const float*)d_in[0];
  const int*   eidx  = (const int*)  d_in[1];
  const int*   batch = (const int*)  d_in[2];
  const float* W1    = (const float*)d_in[3];
  const float* b1    = (const float*)d_in[4];
  const float* W2    = (const float*)d_in[5];
  const float* b2    = (const float*)d_in[6];
  const float* W3    = (const float*)d_in[7];
  const float* b3    = (const float*)d_in[8];
  const float* lin_w = (const float*)d_in[9];
  const float* lin_b = (const float*)d_in[10];
  float* out = (float*)d_out;

  const int N = in_sizes[0] / HIDDEN;   // 100000
  const int E = in_sizes[1] / 2;        // 1600000
  const int G = out_size;               // 64
  const int* erow = eidx;               // edge_index[0]
  const int* ecol = eidx + E;           // edge_index[1]

  // workspace layout
  char* ws = (char*)d_ws;
  size_t off = 0;
  float* dinv = (float*)(ws + off);  off += align256((size_t)N * sizeof(float));
  float* HB0  = (float*)(ws + off);  off += align256((size_t)N * HIDDEN * sizeof(float));
  float* HB1  = (float*)(ws + off);  off += align256((size_t)N * HIDDEN * sizeof(float));
  float* pooled = (float*)(ws + off); off += align256((size_t)G * HIDDEN * sizeof(float));
  float* cnt  = (float*)(ws + off);  off += align256((size_t)G * sizeof(float));
  (void)ws_size; (void)n_in;

  const int TB = 256;
  const int ntiles  = (N + 15) / 16;
  const int gBlocks = (ntiles + 7) / 8;
  const int nvBlocks = (int)(((long long)N * 32 + TB - 1) / TB);   // N*32 lanes
  const int evBlocks = (int)(((long long)E * 32 + TB - 1) / TB);   // E*32 lanes

  // degree -> dinv
  k_deg_init<<<(N + TB - 1) / TB, TB, 0, stream>>>(dinv, N);
  k_deg_accum<<<(E + TB - 1) / TB, TB, 0, stream>>>(ecol, dinv, E);
  k_rsqrt_inplace<<<(N + TB - 1) / TB, TB, 0, stream>>>(dinv, N);

  // layer 1: x -> HB0 (GEMM) -> HB1 (aggregate) -> relu in place
  k_gemm_bf16<<<gBlocks, TB, 0, stream>>>(x, W1, HB0, N, ntiles);
  k_selfloop_init<<<nvBlocks, TB, 0, stream>>>(dinv, HB0, HB1, N);
  k_edge_scatter<<<evBlocks, TB, 0, stream>>>(erow, ecol, dinv, HB0, HB1, E);
  k_bias_relu<<<nvBlocks, TB, 0, stream>>>(HB1, b1, N);

  // layer 2: HB1 -> HB0 (GEMM) -> HB1 (aggregate; GEMM already consumed it)
  k_gemm_bf16<<<gBlocks, TB, 0, stream>>>(HB1, W2, HB0, N, ntiles);
  k_selfloop_init<<<nvBlocks, TB, 0, stream>>>(dinv, HB0, HB1, N);
  k_edge_scatter<<<evBlocks, TB, 0, stream>>>(erow, ecol, dinv, HB0, HB1, E);
  k_bias_relu<<<nvBlocks, TB, 0, stream>>>(HB1, b2, N);

  // layer 3
  k_gemm_bf16<<<gBlocks, TB, 0, stream>>>(HB1, W3, HB0, N, ntiles);
  k_selfloop_init<<<nvBlocks, TB, 0, stream>>>(dinv, HB0, HB1, N);
  k_edge_scatter<<<evBlocks, TB, 0, stream>>>(erow, ecol, dinv, HB0, HB1, E);
  k_bias_relu<<<nvBlocks, TB, 0, stream>>>(HB1, b3, N);

  // mean pool + head
  k_zero<<<(G * HIDDEN + G + TB - 1) / TB, TB, 0, stream>>>(pooled, G * HIDDEN + G);
  k_pool_accum<<<nvBlocks, TB, 0, stream>>>(batch, HB1, pooled, cnt, N);
  k_head<<<G, 128, 0, stream>>>(pooled, cnt, lin_w, lin_b, out);
}
